// DeformableAttention3D_15187004358854
// MI455X (gfx1250) — compile-verified, matrix operand order checked by reading, not run on hardware
//
#include <hip/hip_runtime.h>
#include <hip/hip_bf16.h>

// ---------------- problem constants (from reference) ----------------
#define BS    2
#define NQ    4096
#define CH    256       // C
#define VDIM  48        // X = Y = Z
#define SVOX  (VDIM*VDIM*VDIM)   // 110592
#define NH    8         // heads
#define NP    5         // points
#define HD    32        // head dim = C / H

typedef __bf16 bf16_t;
typedef __attribute__((ext_vector_type(16))) __bf16 v16bf;
typedef __attribute__((ext_vector_type(8)))  __bf16 v8bf;
typedef __attribute__((ext_vector_type(2)))  __bf16 v2bf;
typedef __attribute__((ext_vector_type(8)))  float  v8f;

static __device__ inline unsigned pack_bf16x2(float a, float b) {
  v2bf t; t[0] = (__bf16)a; t[1] = (__bf16)b;   // -> v_cvt_pk_bf16_f32
  return __builtin_bit_cast(unsigned, t);
}

// ---------------------------------------------------------------
// WMMA bf16 fragment loaders. CDNA5 16-bit A-matrix 16x32 layout:
//   lanes 0-15 : row M = lane,     K = {k0+0..7,  k0+16..23}
//   lanes16-31 : row M = lane-16,  K = {k0+8..15, k0+24..31}
// Same striping used for the B (N x K, row-major weights) operand.
// ---------------------------------------------------------------
static __device__ inline v16bf frag_rowmajor(const bf16_t* __restrict__ base,
                                             int ld, int k0, int lane) {
  const int r  = lane & 15;
  const int kk = k0 + ((lane >> 4) << 3);
  const bf16_t* p = base + (size_t)r * (size_t)ld + kk;
  v8bf lo = *reinterpret_cast<const v8bf*>(p);
  v8bf hi = *reinterpret_cast<const v8bf*>(p + 16);
  return __builtin_shufflevector(lo, hi, 0,1,2,3,4,5,6,7,8,9,10,11,12,13,14,15);
}

static __device__ inline v16bf frag_rowmajor_guard(const bf16_t* __restrict__ W,
                                                   int ld, int r0, int k0,
                                                   int lane, int rmax) {
  v16bf f{};
  const int r = r0 + (lane & 15);
  if (r < rmax) {
    const int kk = k0 + ((lane >> 4) << 3);
    const bf16_t* p = W + (size_t)r * (size_t)ld + kk;
    v8bf lo = *reinterpret_cast<const v8bf*>(p);
    v8bf hi = *reinterpret_cast<const v8bf*>(p + 16);
    f = __builtin_shufflevector(lo, hi, 0,1,2,3,4,5,6,7,8,9,10,11,12,13,14,15);
  }
  return f;
}

// ---------------------------------------------------------------
// Elementwise prep: q = bf16(query + pos); weight fp32 -> bf16 casts
// ---------------------------------------------------------------
__global__ void add_cast_kernel(const float* __restrict__ a,
                                const float* __restrict__ b,
                                bf16_t* __restrict__ o, int n) {
  int i = blockIdx.x * blockDim.x + threadIdx.x;
  if (i < n) o[i] = (bf16_t)(a[i] + b[i]);
}

__global__ void cast_kernel(const float* __restrict__ a,
                            bf16_t* __restrict__ o, int n) {
  int i = blockIdx.x * blockDim.x + threadIdx.x;
  if (i < n) o[i] = (bf16_t)a[i];
}

// ---------------------------------------------------------------
// Value projection GEMM, fused transpose through LDS.
//   A[m=s, k=c] = value[b][c][s]  (c-major in memory -> staged+cast in LDS,
//                                  packed bf16x2 ds_store_b32 writes)
//   B[n=d, k=c] = W_val[d][c]     (staged once to LDS, pitch-264 so the
//                                  16B ds_load_b128 fragments are
//                                  bank-conflict-free across lanes)
//   out: v_bf[b][s][d]  (bf16, channel-contiguous for the sampler)
// Block = 256 threads = 8 waves; block tile 128(M) x 64(N); wave tile 16x64.
// Inner loop is pure LDS + WMMA; all B fragments are hoisted ahead of the
// WMMA burst so a single s_wait_dscnt covers the loads and the 4 WMMAs
// issue back-to-back.
// ---------------------------------------------------------------
#define LDS_PITCH 40    // bf16 elems per staged A row (16B-aligned frags)
#define B_PITCH   264   // bf16 elems per staged B row (conflict-free frags)

__global__ __launch_bounds__(256) void value_proj_kernel(
    const float* __restrict__ value, const bf16_t* __restrict__ Wv,
    const float* __restrict__ bias, bf16_t* __restrict__ vout) {
  __shared__ bf16_t ldsA[128 * LDS_PITCH];   // 10240 B
  __shared__ bf16_t ldsB[64 * B_PITCH];      // 33792 B

  const int lane   = threadIdx.x & 31;
  const int wave   = threadIdx.x >> 5;
  const int b      = blockIdx.z;
  const int s0     = blockIdx.x * 128;
  const int n_base = blockIdx.y * 64;

  const float* vbase = value + (size_t)b * CH * (size_t)SVOX;

  // Stage the 64x256 bf16 weight tile once (8 x b128 per thread).
  for (int idx = threadIdx.x; idx < 64 * 32; idx += 256) {
    const int r  = idx >> 5;        // 0..63
    const int ck = (idx & 31) * 8;  // k chunk
    *reinterpret_cast<v8bf*>(&ldsB[r * B_PITCH + ck]) =
        *reinterpret_cast<const v8bf*>(Wv + (size_t)(n_base + r) * CH + ck);
  }

  v8f acc[4] = {};

  for (int kk = 0; kk < CH; kk += 32) {
    __syncthreads();   // protects ldsB on iter 0, ldsA reuse afterwards
    // Stage 128(s) x 32(c) A tile: transpose c-major -> s-major, fp32->bf16,
    // packed pairs of adjacent channels -> ds_store_b32.
    {
      const int sLoc  = (threadIdx.x & 31) * 4;     // coalesced along s
      const int cPair = (threadIdx.x >> 5) * 2;     // 0,2,..,14
#pragma unroll
      for (int i = 0; i < 2; ++i) {
        const int c = cPair + 16 * i;
        const float4 d0 = *reinterpret_cast<const float4*>(
            vbase + (size_t)(kk + c) * SVOX + s0 + sLoc);
        const float4 d1 = *reinterpret_cast<const float4*>(
            vbase + (size_t)(kk + c + 1) * SVOX + s0 + sLoc);
        *reinterpret_cast<unsigned*>(&ldsA[(sLoc + 0) * LDS_PITCH + c]) = pack_bf16x2(d0.x, d1.x);
        *reinterpret_cast<unsigned*>(&ldsA[(sLoc + 1) * LDS_PITCH + c]) = pack_bf16x2(d0.y, d1.y);
        *reinterpret_cast<unsigned*>(&ldsA[(sLoc + 2) * LDS_PITCH + c]) = pack_bf16x2(d0.z, d1.z);
        *reinterpret_cast<unsigned*>(&ldsA[(sLoc + 3) * LDS_PITCH + c]) = pack_bf16x2(d0.w, d1.w);
      }
      if (kk + 32 < CH)   // pull next K-stage toward near caches
        __builtin_prefetch(vbase + (size_t)(kk + 32 + cPair) * SVOX + s0 + sLoc, 0, 3);
    }
    __syncthreads();

    // A fragment from LDS (2x ds_load_b128, conflict-free: pitch 40)
    const int row   = wave * 16 + (lane & 15);
    const int kfrag = (lane >> 4) << 3;
    const bf16_t* pa = &ldsA[row * LDS_PITCH + kfrag];
    v8bf lo = *reinterpret_cast<const v8bf*>(pa);
    v8bf hi = *reinterpret_cast<const v8bf*>(pa + 16);
    v16bf a = __builtin_shufflevector(lo, hi, 0,1,2,3,4,5,6,7,8,9,10,11,12,13,14,15);

    // Hoist all 4 B fragments (8x ds_load_b128) before the WMMA burst.
    v16bf bw[4];
#pragma unroll
    for (int t = 0; t < 4; ++t) {
      const bf16_t* pb = &ldsB[(t * 16 + (lane & 15)) * B_PITCH + kk + kfrag];
      v8bf blo = *reinterpret_cast<const v8bf*>(pb);
      v8bf bhi = *reinterpret_cast<const v8bf*>(pb + 16);
      bw[t] = __builtin_shufflevector(blo, bhi, 0,1,2,3,4,5,6,7,8,9,10,11,12,13,14,15);
    }
#pragma unroll
    for (int t = 0; t < 4; ++t) {
      acc[t] = __builtin_amdgcn_wmma_f32_16x16x32_bf16(
          false, a, false, bw[t], (short)0, acc[t], false, false);
    }
  }

  // Epilogue: 16x16 f32 C/D layout -> bf16 channel-contiguous store
  const int rb = wave * 16 + ((lane >> 4) << 3);
#pragma unroll
  for (int t = 0; t < 4; ++t) {
    const int col = n_base + t * 16 + (lane & 15);
    const float bv = bias[col];
#pragma unroll
    for (int j = 0; j < 8; ++j) {
      const size_t srow = (size_t)s0 + rb + j;
      vout[((size_t)b * SVOX + srow) * CH + col] = (bf16_t)(acc[t][j] + bv);
    }
  }
}

// ---------------------------------------------------------------
// Generic GEMM: D[M,Nreal] = A[M,256](bf16,row-major) x W[Nreal,256]^T + bias
// Output fp32 (ldd-padded) or bf16. Block 128(M) x 64(N), wave 16x64.
// A rows / W rows are L2-hot; direct per-lane 16B fragment loads,
// hoisted ahead of the WMMA burst.
// ---------------------------------------------------------------
__global__ __launch_bounds__(256) void gemm_rowmajor_kernel(
    const bf16_t* __restrict__ A, const bf16_t* __restrict__ W,
    const float* __restrict__ bias, float* __restrict__ Df,
    bf16_t* __restrict__ Dbf, int M, int Nreal, int ldd) {
  const int K = CH;
  const int lane = threadIdx.x & 31;
  const int wave = threadIdx.x >> 5;
  const int m0 = blockIdx.x * 128 + wave * 16;
  const int n_base = blockIdx.y * 64;
  if (m0 >= M) return;   // wave-uniform; EXEC stays full for live waves

  v8f acc[4] = {};
  const bf16_t* Arow = A + (size_t)m0 * K;

  for (int kk = 0; kk < K; kk += 32) {
    v16bf a = frag_rowmajor(Arow, K, kk, lane);
    v16bf bw[4];
#pragma unroll
    for (int t = 0; t < 4; ++t)
      bw[t] = frag_rowmajor_guard(W, K, n_base + t * 16, kk, lane, Nreal);
#pragma unroll
    for (int t = 0; t < 4; ++t)
      acc[t] = __builtin_amdgcn_wmma_f32_16x16x32_bf16(
          false, a, false, bw[t], (short)0, acc[t], false, false);
  }

  const int rbase = m0 + ((lane >> 4) << 3);
#pragma unroll
  for (int t = 0; t < 4; ++t) {
    const int col = n_base + t * 16 + (lane & 15);
    if (col >= Nreal) continue;
    const float bv = bias ? bias[col] : 0.0f;
#pragma unroll
    for (int j = 0; j < 8; ++j) {
      const size_t idx = (size_t)(rbase + j) * ldd + col;
      if (Dbf) Dbf[idx] = (bf16_t)(acc[t][j] + bv);
      else     Df[idx]  = acc[t][j] + bv;
    }
  }
}

// ---------------------------------------------------------------
// Sampling: one wave per (b, query, head); lane = channel within head.
// Softmax over P, trilinear gather from bf16 volume v[b][voxel][256]
// (each corner = one contiguous 64B run across the 32 lanes; the 113MB
// bf16 volume is near-resident in the 192MB L2).
// pix = loc*size - 0.5  (align_corners=False algebra folded).
// ---------------------------------------------------------------
__global__ __launch_bounds__(256) void sample_kernel(
    const bf16_t* __restrict__ v, const float* __restrict__ off_raw,
    const float* __restrict__ aw_raw, const float* __restrict__ refp,
    bf16_t* __restrict__ out_pre) {
  const int lane = threadIdx.x & 31;
  const int gw   = blockIdx.x * 8 + (threadIdx.x >> 5);
  const int h = gw & (NH - 1);
  const int n = (gw >> 3) & (NQ - 1);
  const int b = gw >> 15;            // gw / (NH*NQ)
  const size_t row = (size_t)b * NQ + n;

  // softmax over P (wave-uniform scalars)
  float l[NP];
  float mx = -1e30f;
#pragma unroll
  for (int p = 0; p < NP; ++p) {
    l[p] = aw_raw[row * 48 + h * NP + p];
    mx = fmaxf(mx, l[p]);
  }
  float sum = 0.f;
#pragma unroll
  for (int p = 0; p < NP; ++p) { l[p] = __expf(l[p] - mx); sum += l[p]; }
  const float inv = 1.0f / sum;

  const float rx = refp[row * 3 + 0];
  const float ry = refp[row * 3 + 1];
  const float rz = refp[row * 3 + 2];

  const bf16_t* vb = v + (size_t)b * SVOX * CH + h * HD + lane;
  const float invdim = 1.0f / (float)VDIM;

  float acc = 0.f;
#pragma unroll
  for (int p = 0; p < NP; ++p) {
    const float aw = l[p] * inv;
    const float ox = off_raw[row * 128 + (h * NP + p) * 3 + 0];
    const float oy = off_raw[row * 128 + (h * NP + p) * 3 + 1];
    const float oz = off_raw[row * 128 + (h * NP + p) * 3 + 2];
    const float px = (rx + ox * invdim) * (float)VDIM - 0.5f;
    const float py = (ry + oy * invdim) * (float)VDIM - 0.5f;
    const float pz = (rz + oz * invdim) * (float)VDIM - 0.5f;
    const float lx = floorf(px), ly = floorf(py), lz = floorf(pz);
    const int ix = (int)lx, iy = (int)ly, iz = (int)lz;
    const float fx = px - lx, fy = py - ly, fz = pz - lz;
#pragma unroll
    for (int dx = 0; dx < 2; ++dx)
#pragma unroll
      for (int dy = 0; dy < 2; ++dy)
#pragma unroll
        for (int dz = 0; dz < 2; ++dz) {
          const int xx = ix + dx, yy = iy + dy, zz = iz + dz;
          if ((unsigned)xx < VDIM && (unsigned)yy < VDIM && (unsigned)zz < VDIM) {
            const float w = (dx ? fx : 1.f - fx) * (dy ? fy : 1.f - fy) *
                            (dz ? fz : 1.f - fz);
            const size_t flat = ((size_t)xx * VDIM + yy) * VDIM + zz;
            acc += aw * w * (float)vb[flat * CH];
          }
        }
  }
  out_pre[row * CH + h * HD + lane] = (bf16_t)acc;
}

// ---------------------------------------------------------------
extern "C" void kernel_launch(void* const* d_in, const int* in_sizes, int n_in,
                              void* d_out, int out_size, void* d_ws, size_t ws_size,
                              hipStream_t stream) {
  (void)in_sizes; (void)n_in; (void)out_size; (void)ws_size;
  const float* query  = (const float*)d_in[0];
  const float* value  = (const float*)d_in[1];
  const float* pos    = (const float*)d_in[2];
  const float* refp   = (const float*)d_in[3];
  const float* W_off  = (const float*)d_in[4];
  const float* b_off  = (const float*)d_in[5];
  const float* W_attn = (const float*)d_in[6];
  const float* b_attn = (const float*)d_in[7];
  const float* W_val  = (const float*)d_in[8];
  const float* b_val  = (const float*)d_in[9];
  const float* W_out  = (const float*)d_in[10];
  const float* b_out  = (const float*)d_in[11];
  float* out = (float*)d_out;

  // workspace partitioning (256B aligned)
  char* ws = (char*)d_ws;
  size_t cur = 0;
  auto alloc = [&](size_t bytes) {
    void* p = ws + cur;
    cur = (cur + bytes + 255) & ~(size_t)255;
    return p;
  };
  bf16_t* q_bf    = (bf16_t*)alloc((size_t)BS * NQ * CH * 2);
  bf16_t* Wv_bf   = (bf16_t*)alloc((size_t)CH * CH * 2);
  bf16_t* Wo_bf   = (bf16_t*)alloc((size_t)CH * CH * 2);
  bf16_t* Woff_bf = (bf16_t*)alloc((size_t)120 * CH * 2);
  bf16_t* Wa_bf   = (bf16_t*)alloc((size_t)40 * CH * 2);
  bf16_t* v_bf    = (bf16_t*)alloc((size_t)BS * SVOX * CH * 2);
  float*  off_raw = (float*)alloc((size_t)BS * NQ * 128 * 4);
  float*  aw_raw  = (float*)alloc((size_t)BS * NQ * 48 * 4);
  bf16_t* pre_bf  = (bf16_t*)alloc((size_t)BS * NQ * CH * 2);

  // 1) q = bf16(query + pos); weights -> bf16
  const int nq = BS * NQ * CH;
  add_cast_kernel<<<(nq + 255) / 256, 256, 0, stream>>>(query, pos, q_bf, nq);
  cast_kernel<<<(CH * CH + 255) / 256, 256, 0, stream>>>(W_val, Wv_bf, CH * CH);
  cast_kernel<<<(CH * CH + 255) / 256, 256, 0, stream>>>(W_out, Wo_bf, CH * CH);
  cast_kernel<<<(120 * CH + 255) / 256, 256, 0, stream>>>(W_off, Woff_bf, 120 * CH);
  cast_kernel<<<(40 * CH + 255) / 256, 256, 0, stream>>>(W_attn, Wa_bf, 40 * CH);

  // 2) value projection (fused transpose, bf16 WMMA), v_bf[b][s][c]
  dim3 gv(SVOX / 128, CH / 64, BS);
  value_proj_kernel<<<gv, 256, 0, stream>>>(value, Wv_bf, b_val, v_bf);

  // 3) offset / attention projections (N padded to tile grid, guarded)
  dim3 goff(BS * NQ / 128, 2);
  gemm_rowmajor_kernel<<<goff, 256, 0, stream>>>(q_bf, Woff_bf, b_off,
                                                 off_raw, nullptr, BS * NQ, 120, 128);
  dim3 gattn(BS * NQ / 128, 1);
  gemm_rowmajor_kernel<<<gattn, 256, 0, stream>>>(q_bf, Wa_bf, b_attn,
                                                  aw_raw, nullptr, BS * NQ, 40, 48);

  // 4) softmax + trilinear sampling + P-reduction, one wave per (b,n,h)
  sample_kernel<<<BS * NQ * NH / 8, 256, 0, stream>>>(v_bf, off_raw, aw_raw,
                                                      refp, pre_bf);

  // 5) output projection -> fp32 d_out
  dim3 gout(BS * NQ / 128, CH / 64);
  gemm_rowmajor_kernel<<<gout, 256, 0, stream>>>(pre_bf, Wo_bf, b_out,
                                                 out, nullptr, BS * NQ, CH, CH);
}
